// GraphAttentionLayer_37486474559893
// MI455X (gfx1250) — matrix operand order repaired and checked by reading, not compiled, not run on hardware
//
#include <hip/hip_runtime.h>
#include <hip/hip_bf16.h>
#include <cstdint>

typedef __attribute__((ext_vector_type(16))) _Float16 v16h;
typedef __attribute__((ext_vector_type(8)))  float    v8f;
typedef unsigned int u32x4 __attribute__((ext_vector_type(4)));
typedef int          i32x4 __attribute__((ext_vector_type(4)));
typedef int          i32x8 __attribute__((ext_vector_type(8)));

#define Bsz 4
#define Ssz 1024
#define Dsz 1024
#define Hsz 16
#define HDsz 64

#if __has_builtin(__builtin_amdgcn_tensor_load_to_lds)
#define HAVE_TDM 1
#else
#define HAVE_TDM 0
#endif

__device__ __forceinline__ v8f wmma16(v16h a, v16h b, v8f c) {
  // (neg_a, A, neg_b, B, c_mod, C, reuse_a, reuse_b)
  return __builtin_amdgcn_wmma_f32_16x16x32_f16(false, a, false, b, (short)0, c,
                                                false, false);
}

// A-fragment (16x32 f16, MxK) from LDS row-major tile, row stride `stride` halfs.
// Layout: lane m = lane&15; vgpr v -> K = (lane>=16?16:0) + v*2, K+1
__device__ __forceinline__ v16h frag_a_lds(const _Float16* lds, int row0, int stride) {
  int lane = threadIdx.x & 31;
  int m = lane & 15;
  int koff = (lane >> 4) << 4;
  const _Float16* p = lds + (row0 + m) * stride;
  v16h a;
#pragma unroll
  for (int v = 0; v < 8; ++v) {
    int kb = koff + (v << 1);
    a[2 * v]     = p[kb];
    a[2 * v + 1] = p[kb + 1];
  }
  return a;
}

// B-fragment (32x16 f16, KxN) from LDS tile stored transposed as [N][K] (stride halfs).
// Layout: lane n = lane&15; lanes 0-15 hold K=0..15, lanes 16-31 hold K=16..31; vgpr j -> K=2j,2j+1
__device__ __forceinline__ v16h frag_b_ldsT(const _Float16* lds, int col0, int stride) {
  int lane = threadIdx.x & 31;
  int n = lane & 15;
  int kblk = (lane >> 4) << 4;
  const _Float16* p = lds + (col0 + n) * stride + kblk;
  v16h b;
#pragma unroll
  for (int v = 0; v < 8; ++v) {
    b[2 * v]     = p[2 * v];
    b[2 * v + 1] = p[2 * v + 1];
  }
  return b;
}

#if HAVE_TDM
// Issue one TDM 2D tile load: tile_rows x 32 f16 elements from a [tensor_rows x 1024]
// f16 tensor (row stride 1024 elems) into LDS at lds_off with a 16B pad per 64B row
// (-> 40-half LDS row stride). Descriptor per CDNA5 ISA sec. 8.
__device__ __forceinline__ void tdm_load_tile(unsigned lds_off, const _Float16* gptr,
                                              int tile_rows, int tensor_rows) {
  unsigned long long ga = (unsigned long long)(uintptr_t)(const void*)gptr;
  u32x4 g0 = {
      1u,                                   // count=1 (valid user descriptor)
      lds_off,                              // lds_addr (bytes)
      (unsigned)ga,                         // global_addr[31:0]
      (unsigned)((ga >> 32) & 0x1FFFFFFu) | (2u << 30)  // global_addr[56:32] | type=2
  };
  i32x8 g1 = {
      // data_size=1(2B) | pad_enable | pad_interval=3(16 DW = 64B) | pad_amount=3(4 DW = 16B)
      (int)((1u << 16) | (1u << 20) | (3u << 22) | (3u << 25)),
      (int)(1024u << 16),                   // tensor_dim0[15:0]=1024 in bits 63:48
      (int)((unsigned)tensor_rows << 16),   // tensor_dim1[15:0] in bits 95:80
      (int)(32u << 16),                     // tile_dim0=32 in bits 127:112
      tile_rows,                            // tile_dim1 in bits 143:128 (tile_dim2=0)
      1024,                                 // tensor_dim0_stride[31:0]
      0, 0
  };
  i32x4 z4 = {0, 0, 0, 0};
#if defined(__clang_major__) && (__clang_major__ >= 23)
  i32x8 z8 = {0, 0, 0, 0, 0, 0, 0, 0};
  __builtin_amdgcn_tensor_load_to_lds(g0, g1, z4, z4, z8, 0);
#else
  __builtin_amdgcn_tensor_load_to_lds(g0, g1, z4, z4, 0);
#endif
}
#endif

// ---------------- f32 -> f16 conversion ----------------
__global__ __launch_bounds__(256) void cvt_f32_f16(const float* __restrict__ s,
                                                   _Float16* __restrict__ d, int n4) {
  int i = blockIdx.x * blockDim.x + threadIdx.x;
  if (i < n4) {
    float4 f = ((const float4*)s)[i];
    d[i * 4 + 0] = (_Float16)f.x;
    d[i * 4 + 1] = (_Float16)f.y;
    d[i * 4 + 2] = (_Float16)f.z;
    d[i * 4 + 3] = (_Float16)f.w;
  }
}

// convert + transpose: Wt[c][k] = (f16) W[k][c]   (makes B tiles contiguous in k for TDM)
__global__ __launch_bounds__(256) void cvt_transpose(const float* __restrict__ w,
                                                     _Float16* __restrict__ wt) {
  __shared__ float tile[32][33];
  int c0 = blockIdx.x * 32;
  int k0 = blockIdx.y * 32;
  int tx = threadIdx.x & 31;       // col within tile (coalesced read)
  int ty = threadIdx.x >> 5;       // 8 rows per pass
#pragma unroll
  for (int i = 0; i < 4; ++i)
    tile[ty + i * 8][tx] = w[(size_t)(k0 + ty + i * 8) * Dsz + c0 + tx];
  __syncthreads();
#pragma unroll
  for (int i = 0; i < 4; ++i)
    wt[(size_t)(c0 + ty + i * 8) * Dsz + k0 + tx] = (_Float16)tile[tx][ty + i * 8];
}

// ---------------- QKV projection GEMM (TDM-staged, double-buffered) ----------------
// out[bs, c] = sum_k xh[bs,k] * Wt[c,k] + bias[c]
// grid: (Dsz/128, (B*S)/64, 3); block 256 (8 waves, each a 32x32 subtile)
__global__ __launch_bounds__(256) void qkv_gemm(
    const _Float16* __restrict__ xh,    // [4096,1024]
    const _Float16* __restrict__ wt3,   // [3][1024(c)][1024(k)] transposed
    const float* __restrict__ bq, const float* __restrict__ bk,
    const float* __restrict__ bv,
    _Float16* __restrict__ qout,  // [B,H,S,HD]
    _Float16* __restrict__ kout,  // [B,H,S,HD]
    _Float16* __restrict__ vt)    // [B,H,HD,S]  (transposed)
{
  __shared__ _Float16 lA[2][64 * 40];
  __shared__ _Float16 lB[2][128 * 40];

  const int mat = blockIdx.z;
  const _Float16* wt = wt3 + (size_t)mat * Dsz * Dsz;
  const float* bias = (mat == 0) ? bq : ((mat == 1) ? bk : bv);
  const int row0 = blockIdx.y * 64;
  const int col0 = blockIdx.x * 128;

  const int tid = threadIdx.x;
  const int lane = tid & 31;
  const int wv = tid >> 5;
  const int wr = (wv & 1) * 32;
  const int wc = (wv >> 1) * 32;

  const _Float16* aBase = xh + (size_t)row0 * Dsz;
  const _Float16* bBase = wt + (size_t)col0 * Dsz;

  v8f c00 = {}, c01 = {}, c10 = {}, c11 = {};

#if HAVE_TDM
  const unsigned lAoff = (unsigned)(uintptr_t)(void*)&lA[0][0];
  const unsigned lBoff = (unsigned)(uintptr_t)(void*)&lB[0][0];
  if (wv == 0) {  // prologue: stage k-tile 0 into buffer 0
    tdm_load_tile(lAoff, aBase, 64, 4096);
    tdm_load_tile(lBoff, bBase, 128, 1024);
  }
  for (int kk = 0; kk < Dsz; kk += 32) {
    const int buf = (kk >> 5) & 1;
    if (wv == 0) {
      if (kk + 32 < Dsz) {  // stage next tile into the other buffer, keep it in flight
        tdm_load_tile(lAoff + (buf ^ 1) * (64 * 40 * 2), aBase + kk + 32, 64, 4096);
        tdm_load_tile(lBoff + (buf ^ 1) * (128 * 40 * 2), bBase + kk + 32, 128, 1024);
        __builtin_amdgcn_s_wait_tensorcnt(2);  // current buffer's pair complete
      } else {
        __builtin_amdgcn_s_wait_tensorcnt(0);
      }
    }
    __syncthreads();  // data for `buf` visible to all waves

    v16h a0 = frag_a_lds(&lA[buf][0], wr, 40);
    v16h a1 = frag_a_lds(&lA[buf][0], wr + 16, 40);
    v16h b0 = frag_b_ldsT(&lB[buf][0], wc, 40);
    v16h b1 = frag_b_ldsT(&lB[buf][0], wc + 16, 40);
    c00 = wmma16(a0, b0, c00);
    c01 = wmma16(a0, b1, c01);
    c10 = wmma16(a1, b0, c10);
    c11 = wmma16(a1, b1, c11);
    __syncthreads();  // all waves done with `buf` before it is re-staged
  }
#else
  for (int kk = 0; kk < Dsz; kk += 32) {
#pragma unroll
    for (int i = 0; i < 8; ++i) {   // A tile: 64x32
      int idx = tid + i * 256;
      int r = idx >> 5, k = idx & 31;
      lA[0][r * 40 + k] = aBase[(size_t)r * Dsz + kk + k];
    }
#pragma unroll
    for (int i = 0; i < 16; ++i) {  // B tile: 128x32 from transposed W
      int idx = tid + i * 256;
      int c = idx >> 5, k = idx & 31;
      lB[0][c * 40 + k] = bBase[(size_t)c * Dsz + kk + k];
    }
    __syncthreads();
    v16h a0 = frag_a_lds(&lA[0][0], wr, 40);
    v16h a1 = frag_a_lds(&lA[0][0], wr + 16, 40);
    v16h b0 = frag_b_ldsT(&lB[0][0], wc, 40);
    v16h b1 = frag_b_ldsT(&lB[0][0], wc + 16, 40);
    c00 = wmma16(a0, b0, c00);
    c01 = wmma16(a0, b1, c01);
    c10 = wmma16(a1, b0, c10);
    c11 = wmma16(a1, b1, c11);
    __syncthreads();
  }
#endif

  const int n = lane & 15;
  const int half8 = (lane >> 4) * 8;
  const v8f* accs[4] = {&c00, &c01, &c10, &c11};
#pragma unroll
  for (int t = 0; t < 4; ++t) {
    int rb = (t >> 1) * 16;
    int cb = (t & 1) * 16;
    int col = col0 + wc + cb + n;
    float bsv = bias[col];
    int hh = col >> 6, dd = col & 63;
#pragma unroll
    for (int r = 0; r < 8; ++r) {
      int row = row0 + wr + rb + r + half8;
      int bI = row >> 10, sI = row & 1023;
      float val = (*accs[t])[r] + bsv;
      size_t bh = (size_t)(bI * Hsz + hh);
      if (mat == 0)
        qout[(bh * Ssz + sI) * HDsz + dd] = (_Float16)val;
      else if (mat == 1)
        kout[(bh * Ssz + sI) * HDsz + dd] = (_Float16)val;
      else
        vt[(bh * HDsz + dd) * Ssz + sI] = (_Float16)val;
    }
  }
}

// ---------------- flash attention (one wave per 16-query tile) ----------------
// grid: (S/64, H, B); block 128 (4 waves, each an independent q-tile)
__global__ __launch_bounds__(128) void attn_kernel(
    const _Float16* __restrict__ Q,   // [B,H,S,HD]
    const _Float16* __restrict__ K,   // [B,H,S,HD]
    const _Float16* __restrict__ Vt,  // [B,H,HD,S]
    const float* __restrict__ adj,    // [B,S,S]
    float* __restrict__ out)          // [B,S,D]
{
  __shared__ _Float16 plds[4][16 * 40];

  const int lane = threadIdx.x & 31;
  const int wv = threadIdx.x >> 5;
  const int q0 = (blockIdx.x * 4 + wv) * 16;
  const int h = blockIdx.y;
  const int b = blockIdx.z;
  const int n = lane & 15;
  const int half = lane >> 4;
  const int kblk = half * 16;

  const size_t bh = (size_t)(b * Hsz + h);

  // Q A-fragments: K-dim = d, split 0..31 / 32..63
  const _Float16* qp = Q + (bh * Ssz + q0 + n) * HDsz;
  v16h aq0, aq1;
#pragma unroll
  for (int v = 0; v < 8; ++v) {
    int kb = half * 16 + (v << 1);
    aq0[2 * v] = qp[kb];       aq0[2 * v + 1] = qp[kb + 1];
    aq1[2 * v] = qp[32 + kb];  aq1[2 * v + 1] = qp[32 + kb + 1];
  }

  v8f o0 = {}, o1 = {}, o2 = {}, o3 = {};
  float mrow[8], lrow[8];
#pragma unroll
  for (int r = 0; r < 8; ++r) { mrow[r] = -1e30f; lrow[r] = 0.0f; }

  const float scale = 0.125f;  // 1/sqrt(HD)
  _Float16* pl = plds[wv];
  const float* adjb = adj + (size_t)b * Ssz * Ssz;
  const _Float16* vbase0 = Vt + (bh * HDsz + n) * Ssz;

  for (int key0 = 0; key0 < Ssz; key0 += 32) {
    // ---- scores: two 16x16 tiles (keys key0..15, key0+16..31), K-dim = 64 ----
    const _Float16* krow0 = K + (bh * Ssz + key0 + n) * HDsz;
    const _Float16* krow1 = krow0 + 16 * HDsz;
    if (key0 + 32 < Ssz)  // warm next K tile (global_prefetch)
      __builtin_prefetch(krow0 + 32 * HDsz, 0, 1);
    v16h bk00 = *(const v16h*)(krow0 + kblk);
    v16h bk01 = *(const v16h*)(krow0 + 32 + kblk);
    v16h bk10 = *(const v16h*)(krow1 + kblk);
    v16h bk11 = *(const v16h*)(krow1 + 32 + kblk);

    v8f s0 = {}, s1 = {};
    s0 = wmma16(aq0, bk00, s0);
    s0 = wmma16(aq1, bk01, s0);
    s1 = wmma16(aq0, bk10, s1);
    s1 = wmma16(aq1, bk11, s1);

    // ---- scale + adjacency mask + online softmax ----
#pragma unroll
    for (int r = 0; r < 8; ++r) {
      int row = q0 + r + half * 8;
      float a0v = adjb[(size_t)row * Ssz + key0 + n];
      float a1v = adjb[(size_t)row * Ssz + key0 + 16 + n];
      float p0 = s0[r] * scale + (a0v < 0.5f ? -10000000.0f : 0.0f);
      float p1 = s1[r] * scale + (a1v < 0.5f ? -10000000.0f : 0.0f);

      float mx = fmaxf(p0, p1);
      mx = fmaxf(mx, __shfl_xor(mx, 1));
      mx = fmaxf(mx, __shfl_xor(mx, 2));
      mx = fmaxf(mx, __shfl_xor(mx, 4));
      mx = fmaxf(mx, __shfl_xor(mx, 8));
      float mn = fmaxf(mrow[r], mx);
      float alpha = __expf(mrow[r] - mn);
      float e0 = __expf(p0 - mn);
      float e1 = __expf(p1 - mn);
      float rs = e0 + e1;
      rs += __shfl_xor(rs, 1);
      rs += __shfl_xor(rs, 2);
      rs += __shfl_xor(rs, 4);
      rs += __shfl_xor(rs, 8);
      lrow[r] = lrow[r] * alpha + rs;
      mrow[r] = mn;
      o0[r] *= alpha; o1[r] *= alpha; o2[r] *= alpha; o3[r] *= alpha;

      // stash P (C-layout) into LDS to re-shape into A-fragment layout
      pl[(r + half * 8) * 40 + n]      = (_Float16)e0;
      pl[(r + half * 8) * 40 + 16 + n] = (_Float16)e1;
    }

    v16h ap = frag_a_lds(pl, 0, 40);  // P: 16x32, K-dim = keys

    // ---- P @ V : V B-fragments contiguous thanks to [B,H,HD,S] layout ----
    const _Float16* vp = vbase0 + key0 + kblk;
    v16h bv0 = *(const v16h*)(vp);
    v16h bv1 = *(const v16h*)(vp + (size_t)16 * Ssz);
    v16h bv2 = *(const v16h*)(vp + (size_t)32 * Ssz);
    v16h bv3 = *(const v16h*)(vp + (size_t)48 * Ssz);
    o0 = wmma16(ap, bv0, o0);
    o1 = wmma16(ap, bv1, o1);
    o2 = wmma16(ap, bv2, o2);
    o3 = wmma16(ap, bv3, o3);
  }

  // ---- epilogue: normalize, ReLU, scatter back to [B,S,D] ----
#pragma unroll
  for (int r = 0; r < 8; ++r) {
    int row = q0 + r + half * 8;
    float inv = 1.0f / lrow[r];
    float* op = out + ((size_t)b * Ssz + row) * Dsz + h * HDsz + n;
    op[0]  = fmaxf(o0[r] * inv, 0.0f);
    op[16] = fmaxf(o1[r] * inv, 0.0f);
    op[32] = fmaxf(o2[r] * inv, 0.0f);
    op[48] = fmaxf(o3[r] * inv, 0.0f);
  }
}

extern "C" void kernel_launch(void* const* d_in, const int* in_sizes, int n_in,
                              void* d_out, int out_size, void* d_ws, size_t ws_size,
                              hipStream_t stream) {
  (void)in_sizes; (void)n_in; (void)out_size; (void)ws_size;
  const float* x   = (const float*)d_in[0];
  const float* adj = (const float*)d_in[1];
  const float* Wq  = (const float*)d_in[2];
  const float* bq  = (const float*)d_in[3];
  const float* Wk  = (const float*)d_in[4];
  const float* bk  = (const float*)d_in[5];
  const float* Wv  = (const float*)d_in[6];
  const float* bv  = (const float*)d_in[7];
  float* out = (float*)d_out;

  char* ws = (char*)d_ws;
  _Float16* xh  = (_Float16*)(ws);                        // 8 MB: x as f16
  _Float16* wt3 = (_Float16*)(ws + ((size_t)8  << 20));   // 6 MB: Wq,Wk,Wv f16, transposed
  _Float16* Qh  = (_Float16*)(ws + ((size_t)14 << 20));   // 8 MB
  _Float16* Kh  = (_Float16*)(ws + ((size_t)22 << 20));   // 8 MB
  _Float16* Vt  = (_Float16*)(ws + ((size_t)30 << 20));   // 8 MB (transposed)

  // f32 -> f16 conversions (+ transpose for the weight matrices)
  {
    int n4 = (Bsz * Ssz * Dsz) / 4;  // 1,048,576
    cvt_f32_f16<<<n4 / 256, 256, 0, stream>>>(x, xh, n4);
    dim3 tgrid(Dsz / 32, Dsz / 32);
    cvt_transpose<<<tgrid, 256, 0, stream>>>(Wq, wt3 + 0 * (size_t)Dsz * Dsz);
    cvt_transpose<<<tgrid, 256, 0, stream>>>(Wk, wt3 + 1 * (size_t)Dsz * Dsz);
    cvt_transpose<<<tgrid, 256, 0, stream>>>(Wv, wt3 + 2 * (size_t)Dsz * Dsz);
  }

  // QKV projections (WMMA GEMM, TDM-staged LDS tiles)
  {
    dim3 grid(Dsz / 128, (Bsz * Ssz) / 64, 3);
    qkv_gemm<<<grid, 256, 0, stream>>>(xh, wt3, bq, bk, bv, Qh, Kh, Vt);
  }

  // Masked flash attention + ReLU (WMMA)
  {
    dim3 grid(Ssz / 64, Hsz, Bsz);
    attn_kernel<<<grid, 128, 0, stream>>>(Qh, Kh, Vt, adj, out);
  }

  // second tuple element: adj passthrough
  hipMemcpyAsync(out + (size_t)Bsz * Ssz * Dsz, adj,
                 (size_t)Bsz * Ssz * Ssz * sizeof(float),
                 hipMemcpyDeviceToDevice, stream);
}
